// STENabide_25469156065489
// MI455X (gfx1250) — matrix operand order repaired and checked by reading, not compiled, not run on hardware
//
#include <hip/hip_runtime.h>
#include <hip/hip_bf16.h>
#include <math.h>

// ---------------------------------------------------------------------------
// Model constants
// ---------------------------------------------------------------------------
#define BB  256      // batch
#define SS  200      // seq len
#define DD  200      // base feature dim
#define HH  5        // heads
#define NNODES (BB*SS)   // 51200

typedef __attribute__((ext_vector_type(16))) _Float16 v16h;
typedef __attribute__((ext_vector_type(8)))  _Float16 v8h;
typedef __attribute__((ext_vector_type(4)))  _Float16 v4h;
typedef __attribute__((ext_vector_type(8)))  float    v8f;
typedef __attribute__((ext_vector_type(4)))  float    v4f;

// ---------------------------------------------------------------------------
// Generic batched WMMA GEMM:
//   C[z] = act( alpha * A[z] @ op(B[z]) + bias )
// TRANSB==0:  B is (N,K) row-major  (C = A @ B^T, the nn.Linear case)
// TRANSB==1:  B is (K,N) row-major  (C = A @ B)
// Batch offset for X in {A,B,C}:  (z/zdiv)*sX1 + (z%zdiv)*sX2
// act: 0=none 1=leaky(0.01) 2=gelu(exact, erf)
// Fast path requires 16B-aligned rows: all lda/ldb used here are multiples
// of 4 floats and all base offsets are multiples of 4 floats.
// ---------------------------------------------------------------------------
#define TM 128
#define TN 128
#define TK 32

template<int TRANSB>
__global__ __launch_bounds__(256)
void gemm_wmma_f16(const float* __restrict__ A, const float* __restrict__ Bm,
                   float* __restrict__ C,
                   int M, int N, int K, int lda, int ldb, int ldc,
                   const float* __restrict__ bias, float alpha, int act,
                   int zdiv, long sA1, long sA2, long sB1, long sB2,
                   long sC1, long sC2)
{
    __shared__ __align__(16) _Float16 sA[TM][TK + 8];
    __shared__ __align__(16) _Float16 sB[TN][TK + 8];

    const int z  = blockIdx.z;
    const long zo = z / zdiv, zi = z % zdiv;
    A  += zo * sA1 + zi * sA2;
    Bm += zo * sB1 + zi * sB2;
    C  += zo * sC1 + zi * sC2;

    const int bm = blockIdx.y * TM;
    const int bn = blockIdx.x * TN;

    const int t    = threadIdx.x;
    const int wid  = t >> 5;
    const int lane = t & 31;
    const int wm   = (wid >> 1) * 32;     // wave M offset: 0/32/64/96
    const int wn   = (wid & 1) * 64;      // wave N offset: 0/64

    const int lrow  = lane & 15;
    const int koffA = (lane >> 4) * 8;    // A frag: K 0-7/16-23 vs 8-15/24-31
    const int koffB = (lane >> 4) * 8;    // B frag: K 0-7/16-23 vs 8-15/24-31

    v8f acc[2][4];
    #pragma unroll
    for (int i = 0; i < 2; ++i)
        #pragma unroll
        for (int j = 0; j < 4; ++j)
            #pragma unroll
            for (int e = 0; e < 8; ++e) acc[i][j][e] = 0.0f;

    for (int k0 = 0; k0 < K; k0 += TK) {
        // prefetch next K tile (global_prefetch_b8 on gfx1250)
        if (k0 + TK < K) {
            const int r = t & 127;
            if (bm + r < M) __builtin_prefetch(A + (long)(bm + r) * lda + (k0 + TK), 0, 1);
            if (TRANSB == 0 && bn + r < N)
                __builtin_prefetch(Bm + (long)(bn + r) * ldb + (k0 + TK), 0, 1);
        }

        // ---- A tile: 128x32 f32 -> f16 LDS -------------------------------
        if (bm + TM <= M && k0 + TK <= K) {
            // fast path: float4 global loads, 8B LDS stores, no predication
            #pragma unroll
            for (int i2 = 0; i2 < (TM * TK) / (4 * 256); ++i2) {   // 4 iters
                int lin = t + 256 * i2;
                int r = lin >> 3, c4 = (lin & 7) << 2;
                const v4f q = *(const v4f*)(A + (long)(bm + r) * lda + (k0 + c4));
                v4h h; h[0] = (_Float16)q[0]; h[1] = (_Float16)q[1];
                       h[2] = (_Float16)q[2]; h[3] = (_Float16)q[3];
                *(v4h*)&sA[r][c4] = h;
            }
        } else {
            #pragma unroll
            for (int i2 = 0; i2 < (TM * TK) / 256; ++i2) {
                int idx = t + 256 * i2;
                int r = idx >> 5, c = idx & 31;
                int gm = bm + r, gk = k0 + c;
                float v = (gm < M && gk < K) ? A[(long)gm * lda + gk] : 0.0f;
                sA[r][c] = (_Float16)v;
            }
        }

        // ---- B tile: stored as [n][k] ------------------------------------
        if (TRANSB == 0) {
            if (bn + TN <= N && k0 + TK <= K) {
                #pragma unroll
                for (int i2 = 0; i2 < (TN * TK) / (4 * 256); ++i2) {
                    int lin = t + 256 * i2;
                    int r = lin >> 3, c4 = (lin & 7) << 2;
                    const v4f q = *(const v4f*)(Bm + (long)(bn + r) * ldb + (k0 + c4));
                    v4h h; h[0] = (_Float16)q[0]; h[1] = (_Float16)q[1];
                           h[2] = (_Float16)q[2]; h[3] = (_Float16)q[3];
                    *(v4h*)&sB[r][c4] = h;
                }
            } else {
                #pragma unroll
                for (int i2 = 0; i2 < (TN * TK) / 256; ++i2) {
                    int idx = t + 256 * i2;
                    int r = idx >> 5, c = idx & 31;
                    int gn = bn + r, gk = k0 + c;
                    float v = (gn < N && gk < K) ? Bm[(long)gn * ldb + gk] : 0.0f;
                    sB[r][c] = (_Float16)v;
                }
            }
        } else {
            if (bn + TN <= N && k0 + TK <= K) {
                // read float4 along n at row k, transpose into sB[n][k]
                #pragma unroll
                for (int i2 = 0; i2 < (TN * TK) / (4 * 256); ++i2) {
                    int lin = t + 256 * i2;
                    int k = lin >> 5, n4 = (lin & 31) << 2;
                    const v4f q = *(const v4f*)(Bm + (long)(k0 + k) * ldb + (bn + n4));
                    sB[n4 + 0][k] = (_Float16)q[0];
                    sB[n4 + 1][k] = (_Float16)q[1];
                    sB[n4 + 2][k] = (_Float16)q[2];
                    sB[n4 + 3][k] = (_Float16)q[3];
                }
            } else {
                #pragma unroll
                for (int i2 = 0; i2 < (TN * TK) / 256; ++i2) {
                    int idx = t + 256 * i2;
                    int c = idx >> 7, r = idx & 127;   // coalesced along n
                    int gn = bn + r, gk = k0 + c;
                    float v = (gn < N && gk < K) ? Bm[(long)gk * ldb + gn] : 0.0f;
                    sB[r][c] = (_Float16)v;
                }
            }
        }
        __syncthreads();

        v16h af[2], bf[4];
        #pragma unroll
        for (int mt = 0; mt < 2; ++mt) {
            const v8h lo = *(const v8h*)&sA[wm + mt * 16 + lrow][koffA];
            const v8h hi = *(const v8h*)&sA[wm + mt * 16 + lrow][koffA + 16];
            af[mt] = __builtin_shufflevector(lo, hi, 0,1,2,3,4,5,6,7,8,9,10,11,12,13,14,15);
        }
        #pragma unroll
        for (int nt = 0; nt < 4; ++nt) {
            const v8h lo = *(const v8h*)&sB[wn + nt * 16 + lrow][koffB];
            const v8h hi = *(const v8h*)&sB[wn + nt * 16 + lrow][koffB + 16];
            bf[nt] = __builtin_shufflevector(lo, hi, 0,1,2,3,4,5,6,7,8,9,10,11,12,13,14,15);
        }
        #pragma unroll
        for (int mt = 0; mt < 2; ++mt)
            #pragma unroll
            for (int nt = 0; nt < 4; ++nt)
                acc[mt][nt] = __builtin_amdgcn_wmma_f32_16x16x32_f16(
                    false, af[mt], false, bf[nt], (short)0, acc[mt][nt], false, false);
        __syncthreads();
    }

    // Epilogue. C/D layout: lanes 0-15 -> M=v, lanes 16-31 -> M=8+v; N=lane&15
    const int rb = (lane >> 4) * 8;
    const int cl = lane & 15;
    #pragma unroll
    for (int mt = 0; mt < 2; ++mt)
        #pragma unroll
        for (int nt = 0; nt < 4; ++nt) {
            int col = bn + wn + nt * 16 + cl;
            if (col >= N) continue;
            float bv = bias ? bias[col] : 0.0f;
            #pragma unroll
            for (int v = 0; v < 8; ++v) {
                int row = bm + wm + mt * 16 + rb + v;
                if (row < M) {
                    float x = acc[mt][nt][v] * alpha + bv;
                    if (act == 1)      x = (x >= 0.f) ? x : 0.01f * x;
                    else if (act == 2) x = 0.5f * x * (1.0f + erff(x * 0.70710678118654752f));
                    C[(long)row * ldc + col] = x;
                }
            }
        }
}

// ---------------------------------------------------------------------------
// Elementwise / reduction kernels
// ---------------------------------------------------------------------------
__global__ void k_fill(float* p, float v, long n) {
    long i = (long)blockIdx.x * 256 + threadIdx.x;
    if (i < n) p[i] = v;
}
__global__ void k_deg_count(const int* __restrict__ dst, float* deg, int E) {
    int i = blockIdx.x * 256 + threadIdx.x;
    if (i < E) atomicAdd(&deg[dst[i]], 1.0f);
}
__global__ void k_rsqrt(float* p, int n) {
    int i = blockIdx.x * 256 + threadIdx.x;
    if (i < n) p[i] = rsqrtf(p[i]);
}
__global__ void k_coef(const int* __restrict__ src, const int* __restrict__ dst,
                       const float* __restrict__ dinv, float* coef, int E) {
    int i = blockIdx.x * 256 + threadIdx.x;
    if (i < E) coef[i] = dinv[src[i]] * dinv[dst[i]];
}
__global__ void k_transpose(const float* __restrict__ in, float* __restrict__ out,
                            int K, int N) {  // in (K,N) -> out (N,K)
    int i = blockIdx.x * 256 + threadIdx.x;
    if (i < K * N) { int n = i / K, k = i - n * K; out[i] = in[k * N + n]; }
}
// block per node: out[n,c] = h[n,c]*dinv[n]^2  (self loop term)
__global__ void k_gcn_self(const float* __restrict__ h, const float* __restrict__ dinv,
                           float* out) {
    long n = blockIdx.x;
    int c = threadIdx.x;
    if (c >= DD) return;
    float d = dinv[n];
    out[n * DD + c] = h[n * DD + c] * d * d;
}
// block per edge: out[dst,c] += h[src,c]*coef[e]
__global__ void k_gcn_scatter(const float* __restrict__ h, const int* __restrict__ src,
                              const int* __restrict__ dst, const float* __restrict__ coef,
                              float* out) {
    int e = blockIdx.x;
    int c = threadIdx.x;
    if (c >= DD) return;
    long s = src[e], d = dst[e];
    atomicAdd(out + d * DD + c, h[s * DD + c] * coef[e]);
}
// block per node: x[n,c] = leaky(x[n,c] + bias[c])
__global__ void k_bias_leaky(float* x, const float* __restrict__ bias) {
    long n = blockIdx.x;
    int c = threadIdx.x;
    if (c >= DD) return;
    float v = x[n * DD + c] + bias[c];
    x[n * DD + c] = (v >= 0.f) ? v : 0.01f * v;
}
// softmax over rows of length len (len <= 256), stride ld
__global__ void k_softmax_rows(float* x, int rows, int len, int ld) {
    int r = blockIdx.x;
    if (r >= rows) return;
    float* p = x + (long)r * ld;
    __shared__ float red[256];
    int t = threadIdx.x;
    float v = (t < len) ? p[t] : -3.0e38f;
    red[t] = v; __syncthreads();
    for (int s = 128; s > 0; s >>= 1) { if (t < s) red[t] = fmaxf(red[t], red[t + s]); __syncthreads(); }
    float m = red[0]; __syncthreads();
    float e = (t < len) ? __expf(v - m) : 0.0f;
    red[t] = e; __syncthreads();
    for (int s = 128; s > 0; s >>= 1) { if (t < s) red[t] += red[t + s]; __syncthreads(); }
    float inv = 1.0f / red[0];
    if (t < len) p[t] = e * inv;
}
// out = LN(a + b) * g + be over rows of width d (one block per row)
__global__ void k_ln_residual(const float* __restrict__ a, const float* __restrict__ b,
                              const float* __restrict__ g, const float* __restrict__ be,
                              float* out, int d) {
    long r = blockIdx.x;
    const float* pa = a + r * d; const float* pb = b + r * d;
    float* po = out + r * d;
    int t = threadIdx.x;
    float s = 0.f, s2 = 0.f;
    for (int c = t; c < d; c += 256) { float v = pa[c] + pb[c]; s += v; s2 += v * v; }
    __shared__ float r1[256], r2[256];
    r1[t] = s; r2[t] = s2; __syncthreads();
    for (int k = 128; k > 0; k >>= 1) { if (t < k) { r1[t] += r1[t + k]; r2[t] += r2[t + k]; } __syncthreads(); }
    float mean = r1[0] / d;
    float var  = r2[0] / d - mean * mean;
    float inv  = rsqrtf(var + 1e-5f);
    for (int c = t; c < d; c += 256) { float v = pa[c] + pb[c]; po[c] = (v - mean) * inv * g[c] + be[c]; }
}
// grid (BB, SS): seq[(s*BB+b)*400 + c] = c<200 ? h1[(b*SS+s)*200+c] : x[(b*SS+s)*200+c-200]
__global__ void k_build_seq1(const float* __restrict__ h1, const float* __restrict__ x,
                             float* seq) {
    int b = blockIdx.x, s = blockIdx.y;
    long n  = (long)b * SS + s;
    long ro = ((long)s * BB + b) * 400;
    for (int c = threadIdx.x; c < 400; c += 256)
        seq[ro + c] = (c < DD) ? h1[n * DD + c] : x[n * DD + (c - DD)];
}
// grid (BB, SS): xc[(b*SS+s)*400 + c] = seqA[(s*BB+b)*400 + c]
__global__ void k_build_xc(const float* __restrict__ seqA, float* xc) {
    int b = blockIdx.x, s = blockIdx.y;
    long ri = ((long)s * BB + b) * 400;
    long ro = ((long)b * SS + s) * 400;
    for (int c = threadIdx.x; c < 400; c += 256) xc[ro + c] = seqA[ri + c];
}
// grid (BB, SS): seq2[(s*BB+b)*600 + c] = c<200 ? h2[n*200+c] : seqA[(s*BB+b)*400 + c-200]
__global__ void k_build_seq2(const float* __restrict__ h2, const float* __restrict__ seqA,
                             float* seq2) {
    int b = blockIdx.x, s = blockIdx.y;
    long n  = (long)b * SS + s;
    long ri = ((long)s * BB + b) * 400;
    long ro = ((long)s * BB + b) * 600;
    for (int c = threadIdx.x; c < 600; c += 256)
        seq2[ro + c] = (c < DD) ? h2[n * DD + c] : seqA[ri + (c - DD)];
}
// grid (BB, SS): xcat[(b*SS+s)*600 + c] = seq2[(s*BB+b)*600 + c]
__global__ void k_build_xcat(const float* __restrict__ seq2, float* xcat) {
    int b = blockIdx.x, s = blockIdx.y;
    long ri = ((long)s * BB + b) * 600;
    long ro = ((long)b * SS + s) * 600;
    for (int c = threadIdx.x; c < 600; c += 256) xcat[ro + c] = seq2[ri + c];
}
// gate[n] = dot(gh[n,:k], w2) + b2 (wave per row)
__global__ void k_gate_dot(const float* __restrict__ gh, const float* __restrict__ w2,
                           const float* __restrict__ b2, float* gate, int n, int k) {
    int row  = blockIdx.x * 8 + (threadIdx.x >> 5);
    int lane = threadIdx.x & 31;
    if (row >= n) return;
    const float* p = gh + (long)row * k;
    float s = 0.f;
    for (int c = lane; c < k; c += 32) s += p[c] * w2[c];
    for (int m = 16; m > 0; m >>= 1) s += __shfl_xor(s, m, 32);
    if (lane == 0) gate[row] = s + b2[0];
}
// out[b,c] = sum_s a[b*S+s] * x[(b*S+s)*d + c]
__global__ void k_ga_wsum(const float* __restrict__ a, const float* __restrict__ x,
                          float* out, int d) {
    int b = blockIdx.y;
    int c = blockIdx.x * 256 + threadIdx.x;
    if (c >= d) return;
    float s = 0.f;
    long base = (long)b * SS;
    for (int t = 0; t < SS; ++t) s += a[base + t] * x[(base + t) * d + c];
    out[(long)b * d + c] = s;
}
// final MLP head, one block per batch element
__global__ void k_head(const float* __restrict__ g1, const float* __restrict__ g2,
                       const float* __restrict__ l1w, const float* __restrict__ l1b,
                       const float* __restrict__ l2w, const float* __restrict__ l2b,
                       const float* __restrict__ l3w, const float* __restrict__ l3b,
                       const float* __restrict__ b1g, const float* __restrict__ b1b,
                       const float* __restrict__ b2g, const float* __restrict__ b2b,
                       float* out) {
    __shared__ float z[1000]; __shared__ float z1[128]; __shared__ float z2[32];
    int b = blockIdx.x, t = threadIdx.x;
    for (int c = t; c < 400; c += 256) z[c]       = g1[(long)b * 400 + c];
    for (int c = t; c < 600; c += 256) z[400 + c] = g2[(long)b * 600 + c];
    __syncthreads();
    const float bnScale = rsqrtf(1.0f + 1e-5f);
    if (t < 128) {
        float s = l1b[t];
        const float* w = l1w + (long)t * 1000;
        for (int c = 0; c < 1000; ++c) s += w[c] * z[c];
        s = s * (b1g[t] * bnScale) + b1b[t];
        z1[t] = (s >= 0.f) ? s : 0.01f * s;
    }
    __syncthreads();
    if (t < 32) {
        float s = l2b[t];
        const float* w = l2w + (long)t * 128;
        for (int c = 0; c < 128; ++c) s += w[c] * z1[c];
        s = s * (b2g[t] * bnScale) + b2b[t];
        z2[t] = (s >= 0.f) ? s : 0.01f * s;
    }
    __syncthreads();
    if (t == 0) {
        float s0 = l3b[0], s1 = l3b[1];
        for (int c = 0; c < 32; ++c) { s0 += l3w[c] * z2[c]; s1 += l3w[32 + c] * z2[c]; }
        float m = fmaxf(s0, s1);
        float e0 = __expf(s0 - m), e1 = __expf(s1 - m);
        float inv = 1.0f / (e0 + e1);
        out[(long)b * 2 + 0] = e0 * inv;
        out[(long)b * 2 + 1] = e1 * inv;
    }
}

// ---------------------------------------------------------------------------
// Host-side helpers
// ---------------------------------------------------------------------------
static inline dim3 grid1d(long n) { return dim3((unsigned)((n + 255) / 256)); }

static void launch_gemm(hipStream_t st, int transb,
                        const float* A, const float* Bm, float* C,
                        int M, int N, int K, int lda, int ldb, int ldc,
                        const float* bias, float alpha, int act,
                        int Z = 1, int zdiv = 1,
                        long sA1 = 0, long sA2 = 0, long sB1 = 0, long sB2 = 0,
                        long sC1 = 0, long sC2 = 0)
{
    dim3 g((N + TN - 1) / TN, (M + TM - 1) / TM, Z), blk(256);
    if (transb == 0)
        gemm_wmma_f16<0><<<g, blk, 0, st>>>(A, Bm, C, M, N, K, lda, ldb, ldc,
                                            bias, alpha, act, zdiv, sA1, sA2, sB1, sB2, sC1, sC2);
    else
        gemm_wmma_f16<1><<<g, blk, 0, st>>>(A, Bm, C, M, N, K, lda, ldb, ldc,
                                            bias, alpha, act, zdiv, sA1, sA2, sB1, sB2, sC1, sC2);
}

struct EncW {
    const float *in_w, *in_b, *out_w, *out_b;
    const float *lin1_w, *lin1_b, *lin2_w, *lin2_b;
    const float *n1_g, *n1_b, *n2_g, *n2_b;
};

static void run_encoder(hipStream_t st, float* seq, int d, const EncW& P,
                        float* qkv, float* scores, float* attn, float* ln1)
{
    const int M = SS * BB, d3 = 3 * d, hd = d / HH;
    // qkv = seq @ in_w^T + in_b
    launch_gemm(st, 0, seq, P.in_w, qkv, M, d3, d, d, d, d3, P.in_b, 1.f, 0);
    // scores[z=(b,h)] = Q @ K^T / sqrt(hd)
    launch_gemm(st, 0, qkv, qkv + d, scores, SS, SS, hd, BB * d3, BB * d3, SS,
                nullptr, 1.0f / sqrtf((float)hd), 0,
                BB * HH, HH, (long)d3, (long)hd, (long)d3, (long)hd,
                (long)HH * SS * SS, (long)SS * SS);
    k_softmax_rows<<<BB * HH * SS, 256, 0, st>>>(scores, BB * HH * SS, SS, SS);
    // o = A @ V   (B given (K,N) => TRANSB=1)
    launch_gemm(st, 1, scores, qkv + 2 * d, attn, SS, hd, SS, SS, BB * d3, BB * d,
                nullptr, 1.f, 0,
                BB * HH, HH, (long)HH * SS * SS, (long)SS * SS,
                (long)d3, (long)hd, (long)d, (long)hd);
    // out proj (reuse scores buffer)
    float* proj = scores;
    launch_gemm(st, 0, attn, P.out_w, proj, M, d, d, d, d, d, P.out_b, 1.f, 0);
    // ln1 = LN(seq + proj)
    k_ln_residual<<<M, 256, 0, st>>>(seq, proj, P.n1_g, P.n1_b, ln1, d);
    // ff hidden = gelu(ln1 @ lin1^T + b)   (reuse qkv buffer)
    float* ffh = qkv;
    launch_gemm(st, 0, ln1, P.lin1_w, ffh, M, 1024, d, d, d, 1024, P.lin1_b, 1.f, 2);
    // ff2 = ffh @ lin2^T + b   (reuse attn buffer)
    float* ff2 = attn;
    launch_gemm(st, 0, ffh, P.lin2_w, ff2, M, d, 1024, 1024, 1024, d, P.lin2_b, 1.f, 0);
    // seq_out = LN(ln1 + ff2)  (overwrite enc input)
    k_ln_residual<<<M, 256, 0, st>>>(ln1, ff2, P.n2_g, P.n2_b, seq, d);
}

// ---------------------------------------------------------------------------
extern "C" void kernel_launch(void* const* d_in, const int* in_sizes, int n_in,
                              void* d_out, int out_size, void* d_ws, size_t ws_size,
                              hipStream_t stream)
{
    (void)n_in; (void)out_size; (void)ws_size;
    const float* x    = (const float*)d_in[0];
    const int*   edge = (const int*)d_in[1];
    const int    Ee   = in_sizes[1] / 2;
    const int*   src  = edge;
    const int*   dst  = edge + Ee;

    int i = 2;
    auto F = [&](void) { return (const float*)d_in[i++]; };
    const float *gcn1_b = F(), *gcn1_W = F();          // W stored (K,N) = (200,200)
    const float *gcn2_b = F(), *gcn2_W = F();          // (400,200)
    EncW e1, e2;
    e1.in_w = F(); e1.in_b = F(); e1.out_w = F(); e1.out_b = F();
    e1.lin1_w = F(); e1.lin1_b = F(); e1.lin2_w = F(); e1.lin2_b = F();
    e1.n1_g = F(); e1.n1_b = F(); e1.n2_g = F(); e1.n2_b = F();
    e2.in_w = F(); e2.in_b = F(); e2.out_w = F(); e2.out_b = F();
    e2.lin1_w = F(); e2.lin1_b = F(); e2.lin2_w = F(); e2.lin2_b = F();
    e2.n1_g = F(); e2.n1_b = F(); e2.n2_g = F(); e2.n2_b = F();
    const float *ga1_w1 = F(), *ga1_b1 = F(), *ga1_w2 = F(), *ga1_b2 = F();
    const float *ga2_w1 = F(), *ga2_b1 = F(), *ga2_w2 = F(), *ga2_b2 = F();
    const float *l1_w = F(), *l1_b = F(), *l2_w = F(), *l2_b = F(), *l3_w = F(), *l3_b = F();
    const float *b1_g = F(), *b1_b = F(), *b2_g = F(), *b2_b = F();
    float* out = (float*)d_out;

    // ---- workspace layout (floats) ----
    float* w = (float*)d_ws;
    size_t off = 0;
    auto alloc = [&](size_t nflt) { float* p = w + off; off += nflt; return p; };
    const long M = NNODES;
    float* Wt1    = alloc(200 * 200);                 // gcn1_W transposed -> (N,K)
    float* Wt2    = alloc(200 * 400);                 // gcn2_W transposed -> (200,400)
    float* dinv   = alloc(M);                         // deg -> rsqrt in place
    float* coef   = alloc((size_t)Ee);
    float* lin    = alloc((size_t)M * DD);            // GCN linear output
    float* agg    = alloc((size_t)M * DD);            // GCN aggregated output
    float* seqA   = alloc((size_t)SS * BB * 400);     // seq1 -> encoder1 output
    float* seq2   = alloc((size_t)SS * BB * 600);     // seq2 -> encoder2 output
    float* qkv    = alloc((size_t)M * 1800);          // qkv (also FF hidden)
    float* scores = alloc((size_t)BB * HH * SS * SS); // scores (also proj)
    float* attn   = alloc((size_t)M * 600);           // attn out (also ff2)
    float* ln1    = alloc((size_t)M * 600);
    float* xc     = alloc((size_t)M * 600);           // xc (400) / xcat (600)
    float* gateh  = alloc((size_t)M * 200);           // gate hidden (128 or 200)
    float* gate   = alloc(M);
    float* g1     = alloc((size_t)BB * 400);
    float* g2     = alloc((size_t)BB * 600);

    // ---- preprocessing ----
    k_transpose<<<grid1d(200 * 200), 256, 0, stream>>>(gcn1_W, Wt1, 200, 200);
    k_transpose<<<grid1d(400 * 200), 256, 0, stream>>>(gcn2_W, Wt2, 400, 200);
    k_fill<<<grid1d(M), 256, 0, stream>>>(dinv, 1.0f, M);        // self loops
    k_deg_count<<<grid1d(Ee), 256, 0, stream>>>(dst, dinv, Ee);
    k_rsqrt<<<grid1d(M), 256, 0, stream>>>(dinv, M);
    k_coef<<<grid1d(Ee), 256, 0, stream>>>(src, dst, dinv, coef, Ee);

    // ---- GCN1: h = x @ W1 ; aggregate ; +bias ; leaky ----
    launch_gemm(stream, 0, x, Wt1, lin, M, DD, DD, DD, DD, DD, nullptr, 1.f, 0);
    k_gcn_self<<<M, 256, 0, stream>>>(lin, dinv, agg);
    k_gcn_scatter<<<Ee, 256, 0, stream>>>(lin, src, dst, coef, agg);
    k_bias_leaky<<<M, 256, 0, stream>>>(agg, gcn1_b);

    // ---- seq1 = concat(g2s(h1), g2s(x)) ; encoder a1 ----
    { dim3 g(BB, SS); k_build_seq1<<<g, 256, 0, stream>>>(agg, x, seqA); }
    run_encoder(stream, seqA, 400, e1, qkv, scores, attn, ln1);

    // ---- xc = s2g(seqA) ; gated attention 1 ----
    { dim3 g(BB, SS); k_build_xc<<<g, 256, 0, stream>>>(seqA, xc); }
    launch_gemm(stream, 0, xc, ga1_w1, gateh, M, 128, 400, 400, 400, 128, ga1_b1, 1.f, 0);
    k_gate_dot<<<grid1d(M * 32L), 256, 0, stream>>>(gateh, ga1_w2, ga1_b2, gate, M, 128);
    k_softmax_rows<<<BB, 256, 0, stream>>>(gate, BB, SS, SS);
    { dim3 g((400 + 255) / 256, BB); k_ga_wsum<<<g, 256, 0, stream>>>(gate, xc, g1, 400); }

    // ---- GCN2 on xc ----
    launch_gemm(stream, 0, xc, Wt2, lin, M, DD, 400, 400, 400, DD, nullptr, 1.f, 0);
    k_gcn_self<<<M, 256, 0, stream>>>(lin, dinv, agg);
    k_gcn_scatter<<<Ee, 256, 0, stream>>>(lin, src, dst, coef, agg);
    k_bias_leaky<<<M, 256, 0, stream>>>(agg, gcn2_b);

    // ---- seq2 = concat(g2s(h2), seqA) ; encoder a2 ----
    { dim3 g(BB, SS); k_build_seq2<<<g, 256, 0, stream>>>(agg, seqA, seq2); }
    run_encoder(stream, seq2, 600, e2, qkv, scores, attn, ln1);

    // ---- xcat ; gated attention 2 ----
    { dim3 g(BB, SS); k_build_xcat<<<g, 256, 0, stream>>>(seq2, xc); }
    launch_gemm(stream, 0, xc, ga2_w1, gateh, M, 200, 600, 600, 600, 200, ga2_b1, 1.f, 0);
    k_gate_dot<<<grid1d(M * 32L), 256, 0, stream>>>(gateh, ga2_w2, ga2_b2, gate, M, 200);
    k_softmax_rows<<<BB, 256, 0, stream>>>(gate, BB, SS, SS);
    { dim3 g((600 + 255) / 256, BB); k_ga_wsum<<<g, 256, 0, stream>>>(gate, xc, g2, 600); }

    // ---- head ----
    k_head<<<BB, 256, 0, stream>>>(g1, g2, l1_w, l1_b, l2_w, l2_b, l3_w, l3_b,
                                   b1_g, b1_b, b2_g, b2_b, out);
}
